// PAM_15169824489907
// MI455X (gfx1250) — matrix operand order repaired and checked by reading, not compiled, not run on hardware
//
#include <hip/hip_runtime.h>
#include <hip/hip_bf16.h>

// Problem constants (fixed by the reference).
#define B_   4
#define C_   512
#define HW_  4096
#define HID_ 64

typedef __attribute__((ext_vector_type(16))) __bf16 v16bf;
typedef __attribute__((ext_vector_type(8)))  float  v8f;

union Frag { v16bf v; uint4 q[2]; };

__device__ __forceinline__ v8f zero8() {
  v8f z = {0.f,0.f,0.f,0.f,0.f,0.f,0.f,0.f};
  return z;
}

__device__ __forceinline__ unsigned short f2bf(float f) {
  unsigned u = __float_as_uint(f);
  u += 0x7FFFu + ((u >> 16) & 1u);   // round-to-nearest-even
  return (unsigned short)(u >> 16);
}

// ---------------------------------------------------------------------------
// Kernel 1: fp32 -> bf16 weight conversion
// ---------------------------------------------------------------------------
__global__ void cvt_bf16_kernel(const float* __restrict__ in,
                                unsigned short* __restrict__ out, int n) {
  int i = blockIdx.x * blockDim.x + threadIdx.x;
  if (i < n) out[i] = f2bf(in[i]);
}

// ---------------------------------------------------------------------------
// Kernel 2: x [B][C][HW] f32  ->  xbT [B][HW][C] bf16  (LDS-tiled transpose)
// ---------------------------------------------------------------------------
__global__ __launch_bounds__(256) void transpose_x_kernel(
    const float* __restrict__ x, unsigned short* __restrict__ xbT) {
  __shared__ float tile[32][33];
  const int b  = blockIdx.z;
  const int n0 = blockIdx.x * 32;
  const int c0 = blockIdx.y * 32;
  const int tx = threadIdx.x, ty = threadIdx.y;
#pragma unroll
  for (int i = 0; i < 4; ++i) {
    int c = c0 + ty + i * 8;
    tile[ty + i * 8][tx] = x[((size_t)b * C_ + c) * HW_ + n0 + tx];
  }
  __syncthreads();
#pragma unroll
  for (int i = 0; i < 4; ++i) {
    int n = n0 + ty + i * 8;
    xbT[((size_t)b * HW_ + n) * C_ + c0 + tx] = f2bf(tile[tx][ty + i * 8]);
  }
}

// ---------------------------------------------------------------------------
// Kernel 3: stage-1 GEMM   D[M=out_ch][N=n] = Wb[M][512] * xbT[b][n][512]^T
// One wave computes a 64x64 output tile; K = 512 in 16 WMMA steps.
// TRANS=1: store transposed as bf16 [n][64]   (for qT / kT)
// TRANS=0: store as bf16 [d][HW]              (for v)
// ---------------------------------------------------------------------------
template <int TRANS>
__global__ __launch_bounds__(32) void gemm1_kernel(
    const unsigned short* __restrict__ Wb,
    const unsigned short* __restrict__ xbT,
    unsigned short* __restrict__ outp) {
  const int lane = threadIdx.x & 31;
  const int n0 = blockIdx.x * 64;
  const int m0 = blockIdx.y * 64;
  const int b  = blockIdx.z;
  const int hi  = (lane & 16) ? 1 : 0;
  const int l15 = lane & 15;

  const unsigned short* xb = xbT + (size_t)b * HW_ * C_;
  unsigned short* op = outp + (TRANS ? (size_t)b * HW_ * HID_
                                     : (size_t)b * C_ * HW_);

  v8f acc[4][4];
#pragma unroll
  for (int i = 0; i < 4; ++i)
#pragma unroll
    for (int j = 0; j < 4; ++j) acc[i][j] = zero8();

  for (int k0 = 0; k0 < C_; k0 += 32) {
    Frag a[4], bf[4];
#pragma unroll
    for (int mt = 0; mt < 4; ++mt) {
      // A layout: row = l15, 16 contiguous K starting at hi*16
      const unsigned short* p = Wb + (size_t)(m0 + mt * 16 + l15) * C_ + k0 + hi * 16;
      a[mt].q[0] = *(const uint4*)(p);
      a[mt].q[1] = *(const uint4*)(p + 8);
    }
#pragma unroll
    for (int nt = 0; nt < 4; ++nt) {
      // B layout: col = l15, 16 contiguous K starting at hi*16
      const unsigned short* p = xb + (size_t)(n0 + nt * 16 + l15) * C_ + k0 + hi * 16;
      bf[nt].q[0] = *(const uint4*)(p);
      bf[nt].q[1] = *(const uint4*)(p + 8);
    }
#pragma unroll
    for (int mt = 0; mt < 4; ++mt)
#pragma unroll
      for (int nt = 0; nt < 4; ++nt)
        acc[mt][nt] = __builtin_amdgcn_wmma_f32_16x16x32_bf16(
            false, a[mt].v, false, bf[nt].v, (short)0, acc[mt][nt], false, false);
  }

  if (TRANS) {
    // qT/kT[n][h]: per lane 8 contiguous h values -> one 16B store
#pragma unroll
    for (int mt = 0; mt < 4; ++mt)
#pragma unroll
      for (int nt = 0; nt < 4; ++nt) {
        unsigned short* p = op + (size_t)(n0 + nt * 16 + l15) * HID_ + m0 + mt * 16 + hi * 8;
        uint4 pk;
        pk.x = (unsigned)f2bf(acc[mt][nt][0]) | ((unsigned)f2bf(acc[mt][nt][1]) << 16);
        pk.y = (unsigned)f2bf(acc[mt][nt][2]) | ((unsigned)f2bf(acc[mt][nt][3]) << 16);
        pk.z = (unsigned)f2bf(acc[mt][nt][4]) | ((unsigned)f2bf(acc[mt][nt][5]) << 16);
        pk.w = (unsigned)f2bf(acc[mt][nt][6]) | ((unsigned)f2bf(acc[mt][nt][7]) << 16);
        *(uint4*)p = pk;
      }
  } else {
    // v[d][m]
#pragma unroll
    for (int mt = 0; mt < 4; ++mt)
#pragma unroll
      for (int nt = 0; nt < 4; ++nt)
#pragma unroll
        for (int r = 0; r < 8; ++r) {
          int d = m0 + mt * 16 + hi * 8 + r;
          op[(size_t)d * HW_ + n0 + nt * 16 + l15] = f2bf(acc[mt][nt][r]);
        }
  }
}

// ---------------------------------------------------------------------------
// Kernel 4: flash attention over keys, fused epilogue gamma*O/rowsum + x.
// Workgroup = 8 waves = one 64-query tile of one batch.
//   phase 1: 8 waves cooperatively compute 64x64 logits (WMMA) -> LDS f32
//   phase 2: scalar online softmax (running rowmax/rowsum) -> probs bf16 LDS
//   phase 3: each wave owns 64 d-channels; rescale O, O += P * V^T (WMMA)
//            with all 16 operand fragments preloaded so WMMAs overlap loads.
// ---------------------------------------------------------------------------
__global__ __launch_bounds__(256) void attn_kernel(
    const unsigned short* __restrict__ qT, const unsigned short* __restrict__ kT,
    const unsigned short* __restrict__ vM, const float* __restrict__ x,
    const float* __restrict__ gamma, float* __restrict__ out) {
  __shared__ float Llds[64][65];            // logits (padded vs bank conflicts)
  __shared__ unsigned short Plds[64][64];   // probs, bf16
  __shared__ float rmax[64], rsum[64], osc[64];

  const int b   = blockIdx.y;
  const int n0  = blockIdx.x * 64;
  const int tid = threadIdx.x;
  const int lane = tid & 31;
  const int w    = tid >> 5;
  const int hi   = (lane & 16) ? 1 : 0;
  const int l15  = lane & 15;

  const unsigned short* q = qT + (size_t)b * HW_ * HID_;   // [n][64]
  const unsigned short* k = kT + (size_t)b * HW_ * HID_;   // [m][64]
  const unsigned short* v = vM + (size_t)b * C_ * HW_;     // [d][4096]

  if (tid < 64) { rmax[tid] = -3.0e38f; rsum[tid] = 0.f; }

  // logit-tile assignment: wave w computes tiles (tn_w, tm0) and (tn_w, tm0+1)
  const int tn_w = w >> 1;
  const int tm0  = (w & 1) * 2;

  // Preload this wave's Q A-fragments (rows n0+tn_w*16.., K = 64 in 2 steps).
  Frag qa[2];
  {
    const unsigned short* p = q + (size_t)(n0 + tn_w * 16 + l15) * HID_ + hi * 16;
    qa[0].q[0] = *(const uint4*)(p);
    qa[0].q[1] = *(const uint4*)(p + 8);
    qa[1].q[0] = *(const uint4*)(p + 32);
    qa[1].q[1] = *(const uint4*)(p + 40);
  }

  v8f acc[4][4];                    // O accumulator: 64 queries x 64 d-channels
#pragma unroll
  for (int i = 0; i < 4; ++i)
#pragma unroll
    for (int j = 0; j < 4; ++j) acc[i][j] = zero8();

  const int d0   = w * 64;          // this wave's d-slice
  const int srow = tid >> 2;        // softmax: 4 threads per query row
  const int sc0  = (tid & 3) * 16;

  __syncthreads();

  for (int m0 = 0; m0 < HW_; m0 += 64) {
    // Prefetch next key-step's K rows and this wave's next V tile
    // (lowers to global_prefetch_b8; speculative, dropped if OOB-translated).
    if (m0 + 64 < HW_) {
      __builtin_prefetch(k + (size_t)(m0 + 64 + lane) * HID_, 0, 3);
      __builtin_prefetch(k + (size_t)(m0 + 96 + lane) * HID_, 0, 3);
      __builtin_prefetch(v + (size_t)(d0 + lane) * HW_ + m0 + 64, 0, 3);
      __builtin_prefetch(v + (size_t)(d0 + 32 + lane) * HW_ + m0 + 64, 0, 3);
    }

    // ---- phase 1: logits ----
#pragma unroll
    for (int i = 0; i < 2; ++i) {
      const int tm = tm0 + i;
      Frag b0, b1;
      const unsigned short* p = k + (size_t)(m0 + tm * 16 + l15) * HID_ + hi * 16;
      b0.q[0] = *(const uint4*)(p);
      b0.q[1] = *(const uint4*)(p + 8);
      b1.q[0] = *(const uint4*)(p + 32);
      b1.q[1] = *(const uint4*)(p + 40);
      v8f dt = zero8();
      dt = __builtin_amdgcn_wmma_f32_16x16x32_bf16(false, qa[0].v, false, b0.v,
                                                   (short)0, dt, false, false);
      dt = __builtin_amdgcn_wmma_f32_16x16x32_bf16(false, qa[1].v, false, b1.v,
                                                   (short)0, dt, false, false);
#pragma unroll
      for (int r = 0; r < 8; ++r)
        Llds[tn_w * 16 + hi * 8 + r][tm * 16 + l15] = dt[r];
    }
    __syncthreads();

    // ---- phase 2: online softmax (no cross-wave reductions needed) ----
    {
      float vals[16];
      float lmax = -3.0e38f;
#pragma unroll
      for (int j = 0; j < 16; ++j) {
        vals[j] = Llds[srow][sc0 + j];
        lmax = fmaxf(lmax, vals[j]);
      }
      lmax = fmaxf(lmax, __shfl_xor(lmax, 1, 32));
      lmax = fmaxf(lmax, __shfl_xor(lmax, 2, 32));
      float pm = rmax[srow];
      float nm = fmaxf(pm, lmax);
      float scale = __expf(pm - nm);
      float ls = 0.f;
#pragma unroll
      for (int j = 0; j < 16; ++j) {
        float pv = __expf(vals[j] - nm);
        ls += pv;
        Plds[srow][sc0 + j] = f2bf(pv);
      }
      ls += __shfl_xor(ls, 1, 32);
      ls += __shfl_xor(ls, 2, 32);
      if ((tid & 3) == 0) {
        rsum[srow] = rsum[srow] * scale + ls;
        rmax[srow] = nm;
        osc[srow]  = scale;
      }
    }
    __syncthreads();

    // ---- phase 3: rescale O, O += P * V^T (all operands preloaded) ----
    // Issue all global V loads first, then LDS P loads; the 32 WMMAs then
    // drain the counters progressively instead of stalling per-td.
    Frag vb[2][4];
#pragma unroll
    for (int ks = 0; ks < 2; ++ks)
#pragma unroll
      for (int td = 0; td < 4; ++td) {
        const unsigned short* vp =
            v + (size_t)(d0 + td * 16 + l15) * HW_ + m0 + ks * 32 + hi * 16;
        vb[ks][td].q[0] = *(const uint4*)(vp);
        vb[ks][td].q[1] = *(const uint4*)(vp + 8);
      }
    Frag pa[2][4];
#pragma unroll
    for (int ks = 0; ks < 2; ++ks)
#pragma unroll
      for (int tn = 0; tn < 4; ++tn) {
        const unsigned short* pp = &Plds[tn * 16 + l15][ks * 32 + hi * 16];
        pa[ks][tn].q[0] = *(const uint4*)(pp);
        pa[ks][tn].q[1] = *(const uint4*)(pp + 8);
      }

    // O *= exp(old_max - new_max), per query row
#pragma unroll
    for (int tn = 0; tn < 4; ++tn) {
      float s[8];
#pragma unroll
      for (int r = 0; r < 8; ++r) s[r] = osc[tn * 16 + hi * 8 + r];
#pragma unroll
      for (int td = 0; td < 4; ++td)
#pragma unroll
        for (int r = 0; r < 8; ++r) acc[tn][td][r] *= s[r];
    }

#pragma unroll
    for (int ks = 0; ks < 2; ++ks)
#pragma unroll
      for (int td = 0; td < 4; ++td)
#pragma unroll
        for (int tn = 0; tn < 4; ++tn)
          acc[tn][td] = __builtin_amdgcn_wmma_f32_16x16x32_bf16(
              false, pa[ks][tn].v, false, vb[ks][td].v, (short)0,
              acc[tn][td], false, false);
    __syncthreads();
  }

  // ---- epilogue: out = gamma * (O / rowsum) + x ----
  const float g = gamma[0];
#pragma unroll
  for (int tn = 0; tn < 4; ++tn) {
    float inv[8];
#pragma unroll
    for (int r = 0; r < 8; ++r) inv[r] = 1.f / rsum[tn * 16 + hi * 8 + r];
#pragma unroll
    for (int td = 0; td < 4; ++td) {
      const int d = d0 + td * 16 + l15;
      const size_t base = ((size_t)b * C_ + d) * HW_ + n0 + tn * 16 + hi * 8;
      union { float4 v4[2]; float f[8]; } xi, oo;
      xi.v4[0] = *(const float4*)(x + base);
      xi.v4[1] = *(const float4*)(x + base + 4);
#pragma unroll
      for (int r = 0; r < 8; ++r)
        oo.f[r] = g * acc[tn][td][r] * inv[r] + xi.f[r];
      *(float4*)(out + base)     = oo.v4[0];
      *(float4*)(out + base + 4) = oo.v4[1];
    }
  }
}

// ---------------------------------------------------------------------------
extern "C" void kernel_launch(void* const* d_in, const int* in_sizes, int n_in,
                              void* d_out, int out_size, void* d_ws, size_t ws_size,
                              hipStream_t stream) {
  const float* x     = (const float*)d_in[0];
  const float* Wq    = (const float*)d_in[1];
  const float* Wk    = (const float*)d_in[2];
  const float* Wv    = (const float*)d_in[3];
  const float* gamma = (const float*)d_in[4];
  float* out = (float*)d_out;

  // Workspace layout (bf16 elements), total ~38.4 MB
  unsigned short* ws   = (unsigned short*)d_ws;
  unsigned short* xbT  = ws;                                  // B*HW*C
  unsigned short* wq_b = xbT + (size_t)B_ * HW_ * C_;         // 64*512
  unsigned short* wk_b = wq_b + 64 * 512;
  unsigned short* wv_b = wk_b + 64 * 512;                     // 512*512
  unsigned short* qTp  = wv_b + 512 * 512;                    // B*HW*64
  unsigned short* kTp  = qTp + (size_t)B_ * HW_ * HID_;
  unsigned short* vp   = kTp + (size_t)B_ * HW_ * HID_;       // B*C*HW

  cvt_bf16_kernel<<<(64 * 512 + 255) / 256, 256, 0, stream>>>(Wq, wq_b, 64 * 512);
  cvt_bf16_kernel<<<(64 * 512 + 255) / 256, 256, 0, stream>>>(Wk, wk_b, 64 * 512);
  cvt_bf16_kernel<<<(512 * 512 + 255) / 256, 256, 0, stream>>>(Wv, wv_b, 512 * 512);

  transpose_x_kernel<<<dim3(HW_ / 32, C_ / 32, B_), dim3(32, 8), 0, stream>>>(x, xbT);

  gemm1_kernel<1><<<dim3(HW_ / 64, 1, B_), 32, 0, stream>>>(wq_b, xbT, qTp);
  gemm1_kernel<1><<<dim3(HW_ / 64, 1, B_), 32, 0, stream>>>(wk_b, xbT, kTp);
  gemm1_kernel<0><<<dim3(HW_ / 64, C_ / 64, B_), 32, 0, stream>>>(wv_b, xbT, vp);

  attn_kernel<<<dim3(HW_ / 64, B_), 256, 0, stream>>>(qTp, kTp, vp, x, gamma, out);
}
